// GroupedQueryAttention_48533130445064
// MI455X (gfx1250) — matrix operand order repaired and checked by reading, not compiled, not run on hardware
//
#include <hip/hip_runtime.h>
#include <hip/hip_bf16.h>

// ---------------------------------------------------------------------------
// GQA attention block for gfx1250 (MI455X):
//   - all matmuls via v_wmma_f32_16x16x32_f16 (f16 in, f32 accumulate)
//   - GEMM tiles staged into LDS by the Tensor Data Mover (tensor_load_to_lds),
//     double-buffered and synchronized with s_wait_tensorcnt
// ---------------------------------------------------------------------------

constexpr int S_   = 2048;
constexpr int E_   = 2048;
constexpr int HQ_  = 32;
constexpr int HKV_ = 8;
constexpr int D_   = 64;          // head dim
constexpr int KVW_ = HKV_ * D_;   // 512

typedef __attribute__((ext_vector_type(16))) _Float16 v16h;
typedef __attribute__((ext_vector_type(8)))  _Float16 v8h;
typedef __attribute__((ext_vector_type(8)))  float    v8f;
typedef __attribute__((ext_vector_type(4)))  unsigned int u32x4;
typedef __attribute__((ext_vector_type(4)))  int i32x4;
typedef __attribute__((ext_vector_type(8)))  int i32x8;

#define HAVE_TDM __has_builtin(__builtin_amdgcn_tensor_load_to_lds)

__device__ __forceinline__ v16h cat8(v8h lo, v8h hi) {
  return __builtin_shufflevector(lo, hi, 0,1,2,3,4,5,6,7,8,9,10,11,12,13,14,15);
}

// A fragment 16x32 f16 (M x K). Lane: m = lane&15, g = lane>>4.
// elements 0..7  -> k = g*8 + e ; elements 8..15 -> k = 16 + g*8 + (e-8)
__device__ __forceinline__ v16h load_frag_a(const _Float16* base, int ld) {
  const int lane = threadIdx.x & 31;
  const int m = lane & 15, g = lane >> 4;
  const _Float16* p = base + (size_t)m * ld + g * 8;
  v8h lo = *(const v8h*)p;
  v8h hi = *(const v8h*)(p + 16);
  return cat8(lo, hi);
}

// B fragment 32x16 f16 (K x N) sourced from an N-major buffer: lane n = lane&15,
// g = lane>>4, elements 0..7 -> k = g*8 + e ; elements 8..15 -> k = 16 + g*8 + (e-8).
__device__ __forceinline__ v16h load_frag_b(const _Float16* baseNt, int ld) {
  const int lane = threadIdx.x & 31;
  const int n = lane & 15, g = lane >> 4;
  const _Float16* p = baseNt + (size_t)n * ld + g * 8;
  v8h lo = *(const v8h*)p;
  v8h hi = *(const v8h*)(p + 16);
  return cat8(lo, hi);
}

__device__ __forceinline__ v8f wmma16(v16h a, v16h b, v8f c) {
  return __builtin_amdgcn_wmma_f32_16x16x32_f16(false, a, false, b, (short)0, c,
                                                false, false);
}

// ---------------------------------------------------------------------------
// TDM: 2D tile (rows x 32 halves, row stride = ld halves) -> LDS, async.
// Descriptor layout per CDNA5 ISA ch.8 (group0 128b, group1 256b; groups 2/3
// zero => 2D tensor). Tracked by TENSORcnt.
// ---------------------------------------------------------------------------
#if HAVE_TDM
__device__ __forceinline__ void tdm_load_tile(_Float16* lds_dst,
                                              const _Float16* gsrc,
                                              unsigned rows, unsigned ld) {
  const unsigned lds_off = (unsigned)(uintptr_t)lds_dst;   // low 32b = LDS offset
  const unsigned long long ga = (unsigned long long)(uintptr_t)gsrc;

  union { unsigned u[4]; u32x4 v; } g0 = {};
  g0.u[0] = 1u;                                            // count=1 (valid D#)
  g0.u[1] = lds_off;                                       // lds_addr (bytes)
  g0.u[2] = (unsigned)(ga & 0xffffffffu);                  // global_addr[31:0]
  g0.u[3] = (unsigned)((ga >> 32) & 0x01ffffffu)           // global_addr[56:32]
          | (2u << 30);                                    // type = 2 ("image")

  union { unsigned u[8]; i32x8 v; } g1 = {};
  g1.u[0] = (1u << 16);                                    // data_size = 2 bytes
  // tensor_dim0 [79:48] = ld ; tensor_dim1 [111:80] = rows
  g1.u[1] = (ld & 0xffffu) << 16;
  g1.u[2] = ((ld >> 16) & 0xffffu) | ((rows & 0xffffu) << 16);
  // tile_dim0 [127:112] = 32 (k chunk, contiguous)
  g1.u[3] = ((rows >> 16) & 0xffffu) | (32u << 16);
  // tile_dim1 [143:128] = rows ; tile_dim2 = 0
  g1.u[4] = rows & 0xffffu;
  // tensor_dim0_stride [207:160] = ld (row stride in elements)
  g1.u[5] = ld;
  g1.u[6] = 0;
  g1.u[7] = 0;

  const i32x4 z4 = {};
#if defined(__clang_major__) && (__clang_major__ >= 23)
  const i32x8 z8 = {};
  __builtin_amdgcn_tensor_load_to_lds(g0.v, g1.v, z4, z4, z8, 0);
#else
  __builtin_amdgcn_tensor_load_to_lds(g0.v, g1.v, z4, z4, 0);
#endif
}
#endif

// ---------------------------------------------------------------------------
// f32 -> f16 convert
// ---------------------------------------------------------------------------
__global__ void to_f16_kernel(const float* __restrict__ src,
                              _Float16* __restrict__ dst, int n) {
  int i = blockIdx.x * blockDim.x + threadIdx.x;
  if (i < n) dst[i] = (_Float16)src[i];
}

// Transpose-convert: src is R x C f32 (row-major); dst is C x R f16.
__global__ __launch_bounds__(256)
void transpose_to_f16(const float* __restrict__ src,
                      _Float16* __restrict__ dst, int R, int C) {
  __shared__ float tile[32][33];
  const int c0 = blockIdx.x * 32, r0 = blockIdx.y * 32;
  const int tx = threadIdx.x, ty = threadIdx.y;   // 32 x 8
#pragma unroll
  for (int i = 0; i < 32; i += 8)
    tile[ty + i][tx] = src[(size_t)(r0 + ty + i) * C + (c0 + tx)];
  __syncthreads();
#pragma unroll
  for (int i = 0; i < 32; i += 8)
    dst[(size_t)(c0 + ty + i) * R + (r0 + tx)] = (_Float16)tile[tx][ty + i];
}

// ---------------------------------------------------------------------------
// GEMM: C[M x N] = A[M x K] * B[K x N], B given transposed (Bt is N x K).
// 256 threads = 8 waves; wave grid 4x2; each wave -> 16 rows x 64 cols.
// A (64x32) and B (128x32) k-chunks are DMA'd into double-buffered LDS by the
// TDM (wave 0 issues, pipelined one chunk ahead), all waves consume from LDS.
// ---------------------------------------------------------------------------
template <bool OUT_F32>
__global__ __launch_bounds__(256)
void gemm_tn_kernel(const _Float16* __restrict__ A,
                    const _Float16* __restrict__ Bt,
                    _Float16* __restrict__ Ch, float* __restrict__ Cf,
                    int M, int N, int K) {
  __shared__ __align__(16) _Float16 Alds[2][64][32];    // [buf][m][k]
  __shared__ __align__(16) _Float16 Blds[2][128][32];   // [buf][n][k]

  const int wave = threadIdx.x >> 5;
  const int lane = threadIdx.x & 31;
  const int wm = wave & 3, wn = wave >> 2;
  const int row0 = blockIdx.y * 64;
  const int col0 = blockIdx.x * 128;

#if HAVE_TDM
  // Prologue: DMA chunk 0 into buffer 0.
  if (wave == 0) {
    tdm_load_tile(&Alds[0][0][0], A  + (size_t)row0 * K, 64,  K);
    tdm_load_tile(&Blds[0][0][0], Bt + (size_t)col0 * K, 128, K);
  }
#endif

  v8f acc[4] = {};
  const int nchunks = K >> 5;
  for (int i = 0; i < nchunks; ++i) {
    const int buf = i & 1;
#if HAVE_TDM
    if (wave == 0) {
      if (i + 1 < nchunks) {
        const int kc = (i + 1) << 5;
        tdm_load_tile(&Alds[buf ^ 1][0][0], A  + (size_t)row0 * K + kc, 64,  K);
        tdm_load_tile(&Blds[buf ^ 1][0][0], Bt + (size_t)col0 * K + kc, 128, K);
        __builtin_amdgcn_s_wait_tensorcnt(2);   // chunk i's pair landed
      } else {
        __builtin_amdgcn_s_wait_tensorcnt(0);
      }
    }
#else
    // Fallback: cooperative vector copy of both tiles.
    {
      const int kc = i << 5;
      const int t = threadIdx.x;
      {
        const int r = t >> 2, kg = (t & 3) * 8;
        *(v8h*)&Alds[buf][r][kg] = *(const v8h*)(A + (size_t)(row0 + r) * K + kc + kg);
      }
#pragma unroll
      for (int rep = 0; rep < 2; ++rep) {
        const int idx = t + rep * 256;
        const int r = idx >> 2, kg = (idx & 3) * 8;
        *(v8h*)&Blds[buf][r][kg] = *(const v8h*)(Bt + (size_t)(col0 + r) * K + kc + kg);
      }
    }
#endif
    __syncthreads();   // tile visible to all waves

    v16h a = load_frag_a(&Alds[buf][wm * 16][0], 32);
#pragma unroll
    for (int t = 0; t < 4; ++t) {
      v16h b = load_frag_b(&Blds[buf][wn * 64 + t * 16][0], 32);
      acc[t] = wmma16(a, b, acc[t]);
    }
    __syncthreads();   // all reads done before this buffer is re-filled
  }

  const int g = lane >> 4, n = lane & 15;
#pragma unroll
  for (int t = 0; t < 4; ++t)
#pragma unroll
    for (int v = 0; v < 8; ++v) {
      const int r = row0 + wm * 16 + v + g * 8;
      const int c = col0 + wn * 64 + t * 16 + n;
      if (OUT_F32) Cf[(size_t)r * N + c] = acc[t][v];
      else         Ch[(size_t)r * N + c] = (_Float16)acc[t][v];
    }
}

// ---------------------------------------------------------------------------
// RoPE in-place on f16 [s][h][d] buffer; freqs_cis is (S, D/2, 2) f32.
// ---------------------------------------------------------------------------
__global__ void rope_kernel(_Float16* __restrict__ t,
                            const float* __restrict__ fc, int H, int total) {
  int idx = blockIdx.x * blockDim.x + threadIdx.x;   // pair index
  if (idx >= total) return;
  const int i = idx & 31;                 // D/2 = 32
  const int h = (idx >> 5) % H;
  const int s = idx / (32 * H);
  _Float16* p = t + ((size_t)s * H + h) * D_ + 2 * i;
  const float fr = fc[((size_t)s * 32 + i) * 2 + 0];
  const float fi = fc[((size_t)s * 32 + i) * 2 + 1];
  const float t0 = (float)p[0], t1 = (float)p[1];
  p[0] = (_Float16)(t0 * fr - t1 * fi);
  p[1] = (_Float16)(t0 * fi + t1 * fr);
}

// ---------------------------------------------------------------------------
// Flash attention. Grid: (S/64, HQ). Block: 128 threads = 4 waves.
// Wave w owns query rows [q0 + 16w, q0 + 16w + 15] of head blockIdx.y.
// Per 32-key chunk: 4 WMMA for scores + 4 WMMA for PV.
// ---------------------------------------------------------------------------
__global__ __launch_bounds__(128)
void attn_kernel(const _Float16* __restrict__ qh, const _Float16* __restrict__ kh,
                 const _Float16* __restrict__ vh, _Float16* __restrict__ oh) {
  __shared__ __align__(16) _Float16 Vt[64][40];     // V chunk transposed: [d][key]
  __shared__ __align__(16) _Float16 Pb[4][16][40];  // per-wave P tile (16 x 32)

  const int h   = blockIdx.y;
  const int hkv = h >> 2;                 // REP = 4
  const int q0  = blockIdx.x * 64;
  const int wave = threadIdx.x >> 5;
  const int lane = threadIdx.x & 31;
  const int qw = q0 + wave * 16;
  const int g = lane >> 4, ln = lane & 15;
  const float scale = 0.125f;             // 1/sqrt(64)

  const _Float16* qbase = qh + (size_t)qw * E_ + h * D_;
  const v16h a0 = load_frag_a(qbase + 0,  E_);
  const v16h a1 = load_frag_a(qbase + 32, E_);

  v8f o[4] = {};
  float mrow[8], lrow[8];
#pragma unroll
  for (int v = 0; v < 8; ++v) { mrow[v] = -1e30f; lrow[v] = 0.f; }

  const int kend = q0 + 64;               // causal: keys <= last query of block
  for (int kc = 0; kc < kend; kc += 32) {
    // --- cooperatively stage V chunk transposed into LDS -------------------
    {
      const int key   = threadIdx.x & 31;          // 0..31
      const int dbase = (threadIdx.x >> 5) * 16;   // 0,16,32,48
      const _Float16* src = vh + (size_t)(kc + key) * KVW_ + hkv * D_ + dbase;
      v8h x0 = *(const v8h*)src;
      v8h x1 = *(const v8h*)(src + 8);
#pragma unroll
      for (int j = 0; j < 8; ++j) Vt[dbase + j][key] = x0[j];
#pragma unroll
      for (int j = 0; j < 8; ++j) Vt[dbase + 8 + j][key] = x1[j];
    }
    __syncthreads();

    // --- scores: S(16x32) = Q(16x64) * K^T(64x32), 4 WMMA ------------------
    v8f s0 = {}, s1 = {};
    {
      const _Float16* kb = kh + hkv * D_;
      v16h b00 = load_frag_b(kb + (size_t)(kc)      * KVW_ + 0,  KVW_);
      v16h b01 = load_frag_b(kb + (size_t)(kc)      * KVW_ + 32, KVW_);
      v16h b10 = load_frag_b(kb + (size_t)(kc + 16) * KVW_ + 0,  KVW_);
      v16h b11 = load_frag_b(kb + (size_t)(kc + 16) * KVW_ + 32, KVW_);
      s0 = wmma16(a0, b00, s0);  s0 = wmma16(a1, b01, s0);
      s1 = wmma16(a0, b10, s1);  s1 = wmma16(a1, b11, s1);
    }

    // --- online softmax in the C layout (row = v + 8g, col = ln) -----------
    float p0[8], p1[8], alpha[8];
#pragma unroll
    for (int v = 0; v < 8; ++v) {
      const int row = qw + v + g * 8;
      const int c0 = kc + ln, c1 = kc + 16 + ln;
      float x0 = s0[v] * scale + ((c0 > row) ? -1e9f : 0.f);
      float x1 = s1[v] * scale + ((c1 > row) ? -1e9f : 0.f);
      float t = fmaxf(x0, x1);
#pragma unroll
      for (int off = 8; off; off >>= 1) t = fmaxf(t, __shfl_xor(t, off, 32));
      const float nm = fmaxf(mrow[v], t);
      alpha[v] = __expf(mrow[v] - nm);
      mrow[v] = nm;
      p0[v] = __expf(x0 - nm);
      p1[v] = __expf(x1 - nm);
      float r = p0[v] + p1[v];
#pragma unroll
      for (int off = 8; off; off >>= 1) r += __shfl_xor(r, off, 32);
      lrow[v] = lrow[v] * alpha[v] + r;
    }
#pragma unroll
    for (int t = 0; t < 4; ++t)
#pragma unroll
      for (int v = 0; v < 8; ++v) o[t][v] *= alpha[v];

    // --- reshape P (C layout -> A layout) through per-wave LDS tile --------
#pragma unroll
    for (int v = 0; v < 8; ++v) {
      Pb[wave][v + g * 8][ln]      = (_Float16)p0[v];
      Pb[wave][v + g * 8][16 + ln] = (_Float16)p1[v];
    }
    const v16h ap = load_frag_a(&Pb[wave][0][0], 40);

    // --- O(16x64) += P(16x32) * V(32x64), 4 WMMA ---------------------------
#pragma unroll
    for (int t = 0; t < 4; ++t) {
      v16h bv = load_frag_b(&Vt[t * 16][0], 40);
      o[t] = wmma16(ap, bv, o[t]);
    }
    __syncthreads();   // protect Vt before next chunk overwrites it
  }

  // --- epilogue: normalize and store f16 -----------------------------------
#pragma unroll
  for (int t = 0; t < 4; ++t)
#pragma unroll
    for (int v = 0; v < 8; ++v) {
      const float val = o[t][v] / lrow[v];
      const int row = qw + v + g * 8;
      oh[(size_t)row * E_ + h * D_ + t * 16 + ln] = (_Float16)val;
    }
}

// ---------------------------------------------------------------------------
// Launch: convert -> project (QKV) -> RoPE -> flash attention -> out proj
// ---------------------------------------------------------------------------
extern "C" void kernel_launch(void* const* d_in, const int* in_sizes, int n_in,
                              void* d_out, int out_size, void* d_ws, size_t ws_size,
                              hipStream_t stream) {
  const float* x  = (const float*)d_in[0];
  const float* fc = (const float*)d_in[1];
  // d_in[2] = mask: causality handled analytically
  const float* Wq = (const float*)d_in[3];
  const float* Wk = (const float*)d_in[4];
  const float* Wv = (const float*)d_in[5];
  const float* Wo = (const float*)d_in[6];
  float* out = (float*)d_out;

  _Float16* ws = (_Float16*)d_ws;
  size_t off = 0;
  _Float16* xh  = ws + off; off += (size_t)S_ * E_;   // x f16 (reused for attn out)
  _Float16* Wqt = ws + off; off += (size_t)E_ * E_;
  _Float16* Wkt = ws + off; off += (size_t)E_ * KVW_;
  _Float16* Wvt = ws + off; off += (size_t)E_ * KVW_;
  _Float16* Wot = ws + off; off += (size_t)E_ * E_;
  _Float16* qh  = ws + off; off += (size_t)S_ * E_;
  _Float16* kh  = ws + off; off += (size_t)S_ * KVW_;
  _Float16* vh  = ws + off; off += (size_t)S_ * KVW_;
  _Float16* ah  = xh;  // attn output aliases xh (x dead after projections)

  // 1) converts / weight transposes
  {
    int n = S_ * E_;
    to_f16_kernel<<<(n + 255) / 256, 256, 0, stream>>>(x, xh, n);
    transpose_to_f16<<<dim3(E_ / 32, E_ / 32), dim3(32, 8), 0, stream>>>(Wq, Wqt, E_, E_);
    transpose_to_f16<<<dim3(KVW_ / 32, E_ / 32), dim3(32, 8), 0, stream>>>(Wk, Wkt, E_, KVW_);
    transpose_to_f16<<<dim3(KVW_ / 32, E_ / 32), dim3(32, 8), 0, stream>>>(Wv, Wvt, E_, KVW_);
    transpose_to_f16<<<dim3(E_ / 32, E_ / 32), dim3(32, 8), 0, stream>>>(Wo, Wot, E_, E_);
  }

  // 2) QKV projections (WMMA + TDM-staged LDS tiles)
  gemm_tn_kernel<false><<<dim3(E_ / 128,   S_ / 64), 256, 0, stream>>>(xh, Wqt, qh, nullptr, S_, E_,   E_);
  gemm_tn_kernel<false><<<dim3(KVW_ / 128, S_ / 64), 256, 0, stream>>>(xh, Wkt, kh, nullptr, S_, KVW_, E_);
  gemm_tn_kernel<false><<<dim3(KVW_ / 128, S_ / 64), 256, 0, stream>>>(xh, Wvt, vh, nullptr, S_, KVW_, E_);

  // 3) RoPE on q and k
  {
    int nq = S_ * HQ_  * (D_ / 2);
    int nk = S_ * HKV_ * (D_ / 2);
    rope_kernel<<<(nq + 255) / 256, 256, 0, stream>>>(qh, fc, HQ_,  nq);
    rope_kernel<<<(nk + 255) / 256, 256, 0, stream>>>(kh, fc, HKV_, nk);
  }

  // 4) causal flash attention (WMMA for QK^T and PV)
  attn_kernel<<<dim3(S_ / 64, HQ_), 128, 0, stream>>>(qh, kh, vh, ah);

  // 5) output projection, f32 result (WMMA)
  gemm_tn_kernel<true><<<dim3(E_ / 128, S_ / 64), 256, 0, stream>>>(ah, Wot, nullptr, out, S_, E_, E_);
}
